// CLUB_47794396070568
// MI455X (gfx1250) — matrix operand order repaired and checked by reading, not compiled, not run on hardware
//
#include <hip/hip_runtime.h>
#include <hip/hip_bf16.h>

// ---------------------------------------------------------------------------
// CLUB MI estimator, CDNA5 (gfx1250) wave32 + v_wmma_f32_16x16x32_bf16
//   mu      = relu(z_c@W1mu + b1mu)@W2mu + b2mu
//   logvar  = tanh(relu(z_c@W1lv + b1lv)@W2lv + b2lv)
//   mi      = mean_i sum_d inv2var*((Ezd2 - 2 mu Ezd + mu^2) - (mu - z_d)^2)
// N=16384, DC=H=DD=1024.  ~137 GFLOP of GEMM -> bf16 WMMA, f32 accumulate.
// GEMM staging: async global->LDS (ASYNCcnt) with 2-stage double buffering
// when the toolchain exposes the builtin; sync ds_store path otherwise.
// ---------------------------------------------------------------------------

#define N_ROWS 16384
#define KDIM   1024      // DC == H == DD == 1024
#define BM     128
#define BN     128
#define BK     32
#define NKT    (KDIM / BK)
#define LDT    48        // padded LDS row stride (bf16 units): 96B, 16B-aligned
#define TSZ    (BM * LDT) // one tile buffer in bf16 units

#if defined(__has_builtin)
#if __has_builtin(__builtin_amdgcn_global_load_async_to_lds_b128)
#define CLUB_ASYNC 1
#endif
#endif

typedef __attribute__((ext_vector_type(16))) __bf16 v16bf;
typedef __attribute__((ext_vector_type(8)))  float  v8f;
typedef int v4i __attribute__((vector_size(16)));
typedef __attribute__((address_space(1))) v4i* gv4i_ptr;   // global (AS1)
typedef __attribute__((address_space(3))) v4i* lv4i_ptr;   // LDS    (AS3)

union FragBF { v16bf v; uint4 u[2]; };

__device__ __forceinline__ unsigned short f32_to_bf16(float f) {
    unsigned u = __float_as_uint(f);
    unsigned r = (u + 0x7FFFu + ((u >> 16) & 1u)) >> 16;   // round-nearest-even
    return (unsigned short)r;
}

// ------------------------------- small kernels ------------------------------

__global__ void club_zero_kernel(float* out, float* sumd, float* sumsq) {
    int t = threadIdx.x;
    if (t == 0) out[0] = 0.0f;
    for (int c = t; c < KDIM; c += 256) { sumd[c] = 0.0f; sumsq[c] = 0.0f; }
}

// f32 -> bf16, vectorized (float4 in, 4x u16 out)
__global__ void club_cvt_bf16_kernel(const float* __restrict__ in,
                                     unsigned short* __restrict__ out, int n4) {
    int i = blockIdx.x * blockDim.x + threadIdx.x;
    if (i >= n4) return;
    float4 v = ((const float4*)in)[i];
    ushort4 o;
    o.x = f32_to_bf16(v.x); o.y = f32_to_bf16(v.y);
    o.z = f32_to_bf16(v.z); o.w = f32_to_bf16(v.w);
    ((ushort4*)out)[i] = o;
}

// W [K][N] f32 row-major  ->  Wt [N][K] bf16 (K-contiguous for WMMA B frags)
__global__ void club_cvt_transpose_kernel(const float* __restrict__ in,
                                          unsigned short* __restrict__ out) {
    int idx = blockIdx.x * blockDim.x + threadIdx.x;       // grid covers K*N
    int k = idx >> 10;            // / 1024
    int n = idx & 1023;           // % 1024
    out[(size_t)n * KDIM + k] = f32_to_bf16(in[idx]);
}

// column sums of z_d and z_d^2 (for separable "negative" term)
__global__ __launch_bounds__(256)
void club_colstats_kernel(const float* __restrict__ zd,
                          float* __restrict__ sumd, float* __restrict__ sumsq) {
    int tid = threadIdx.x;
    int r0  = blockIdx.x * 64;                              // 256 blocks x 64 rows
    float s[4] = {0.f,0.f,0.f,0.f}, q[4] = {0.f,0.f,0.f,0.f};
    for (int r = 0; r < 64; ++r) {
        const float* row = zd + (size_t)(r0 + r) * KDIM;
#pragma unroll
        for (int j = 0; j < 4; ++j) {
            float v = row[tid + j * 256];
            s[j] += v; q[j] += v * v;
        }
    }
#pragma unroll
    for (int j = 0; j < 4; ++j) {
        atomicAdd(&sumd [tid + j * 256], s[j]);
        atomicAdd(&sumsq[tid + j * 256], q[j]);
    }
}

// ------------------------------- GEMM core ----------------------------------

#ifdef CLUB_ASYNC
// 128x32 bf16 tile: issue async global->LDS copies (tracked by ASYNCcnt).
__device__ __forceinline__ void club_async_tile(const unsigned short* __restrict__ g,
                                                int row0, int k0,
                                                unsigned short* lds) {
    int tid = threadIdx.x;
#pragma unroll
    for (int j = 0; j < 2; ++j) {
        int c   = tid + j * 256;          // 512 x 16B chunks
        int row = c >> 2;
        int off = (c & 3) * 8;            // bf16 units
        __builtin_amdgcn_global_load_async_to_lds_b128(
            (gv4i_ptr)(g + (size_t)(row0 + row) * KDIM + k0 + off),
            (lv4i_ptr)(lds + row * LDT + off),
            0, 0);
    }
}
#else
// sync fallback: global_load_b128 -> VGPR -> ds_store_b128
__device__ __forceinline__ void club_load_tile(const unsigned short* __restrict__ g,
                                               int row0, int k0,
                                               unsigned short* lds) {
    int tid = threadIdx.x;
#pragma unroll
    for (int j = 0; j < 2; ++j) {
        int c   = tid + j * 256;
        int row = c >> 2;
        int off = (c & 3) * 8;
        uint4 v = *(const uint4*)(g + (size_t)(row0 + row) * KDIM + k0 + off);
        *(uint4*)(lds + row * LDT + off) = v;
    }
}
#endif

// Fragment loads + 8 WMMAs for one K-step of a 64x32 wave tile.
__device__ __forceinline__ void club_wmma_step(const unsigned short* cA,
                                               const unsigned short* cB,
                                               int wm, int wn, int mrow, int hi,
                                               v8f acc[4][2]) {
    // A 16x32 frags: lanes0-15 rows m, K 0-7 & 16-23; lanes16-31: K 8-15 & 24-31
    FragBF a[4];
#pragma unroll
    for (int i = 0; i < 4; ++i) {
        const unsigned short* p = cA + (wm + i * 16 + mrow) * LDT + hi * 8;
        a[i].u[0] = *(const uint4*)p;
        a[i].u[1] = *(const uint4*)(p + 16);
    }
    // B 32x16 frags: lane holds column n; lanes0-15 K 0-7 & 16-23; lanes16-31: K 8-15 & 24-31
    FragBF b[2];
#pragma unroll
    for (int j = 0; j < 2; ++j) {
        const unsigned short* p = cB + (wn + j * 16 + mrow) * LDT + hi * 8;
        b[j].u[0] = *(const uint4*)p;
        b[j].u[1] = *(const uint4*)(p + 16);
    }
#pragma unroll
    for (int i = 0; i < 4; ++i)
#pragma unroll
        for (int j = 0; j < 2; ++j)
            acc[i][j] = __builtin_amdgcn_wmma_f32_16x16x32_bf16(
                false, a[i].v, false, b[j].v, (short)0, acc[i][j], false, false);
}

// One 128x128 block tile, K=1024: acc[4][2] per wave (wave tile 64x32).
// sA/sB are double buffers of 2*TSZ bf16 each.
__device__ __forceinline__ void club_gemm_block(const unsigned short* __restrict__ A,
                                                const unsigned short* __restrict__ Bt,
                                                int M0, int N0,
                                                unsigned short* sA, unsigned short* sB,
                                                v8f acc[4][2]) {
    int tid  = threadIdx.x;
    int lane = tid & 31;
    int wave = tid >> 5;
    int wm   = (wave >> 2) * 64;          // wave M offset in block
    int wn   = (wave & 3) * 32;           // wave N offset in block
    int mrow = lane & 15;
    int hi   = lane >> 4;                 // half-wave select

#ifdef CLUB_ASYNC
    __syncthreads();                      // retire any previous LDS consumers
    club_async_tile(A,  M0, 0, sA);       // prologue: stage tile 0 into buf 0
    club_async_tile(Bt, N0, 0, sB);
    for (int kt = 0; kt < NKT; ++kt) {
        int cur = kt & 1, nxt = cur ^ 1;
        asm volatile("s_wait_asynccnt 0x0" ::: "memory");  // own copies done
        __syncthreads();                                   // all waves' copies visible
        if (kt + 1 < NKT) {               // stage tile kt+1 into the other buffer
            club_async_tile(A,  M0, (kt + 1) * BK, sA + nxt * TSZ);
            club_async_tile(Bt, N0, (kt + 1) * BK, sB + nxt * TSZ);
        }
        club_wmma_step(sA + cur * TSZ, sB + cur * TSZ, wm, wn, mrow, hi, acc);
    }
#else
    for (int k0 = 0; k0 < KDIM; k0 += BK) {
        __syncthreads();
        club_load_tile(A,  M0, k0, sA);
        club_load_tile(Bt, N0, k0, sB);
        if (k0 + BK < KDIM) {             // prefetch next K-tile into near caches
            int pr = tid >> 1;
            __builtin_prefetch(A  + (size_t)(M0 + pr) * KDIM + k0 + BK, 0, 3);
            __builtin_prefetch(Bt + (size_t)(N0 + pr) * KDIM + k0 + BK, 0, 3);
        }
        __syncthreads();
        club_wmma_step(sA, sB, wm, wn, mrow, hi, acc);
    }
#endif
}

// GEMM1: h1 = relu(zc @ W1 + b1), bf16 out.  grid (128, 8), block 256.
__global__ __launch_bounds__(256)
void club_gemm1_relu_kernel(const unsigned short* __restrict__ zc,
                            const unsigned short* __restrict__ w1t,
                            const float* __restrict__ b1,
                            unsigned short* __restrict__ h1) {
    __shared__ __align__(16) unsigned short sA[2 * TSZ];
    __shared__ __align__(16) unsigned short sB[2 * TSZ];
    int M0 = blockIdx.x * BM, N0 = blockIdx.y * BN;

    v8f acc[4][2];
#pragma unroll
    for (int i = 0; i < 4; ++i)
#pragma unroll
        for (int j = 0; j < 2; ++j) acc[i][j] = (v8f)0.0f;

    club_gemm_block(zc, w1t, M0, N0, sA, sB, acc);

    int lane = threadIdx.x & 31, wave = threadIdx.x >> 5;
    int wm = (wave >> 2) * 64, wn = (wave & 3) * 32;
    int mrow = lane & 15, hi = lane >> 4;
#pragma unroll
    for (int i = 0; i < 4; ++i)
#pragma unroll
        for (int j = 0; j < 2; ++j) {
            int n = N0 + wn + j * 16 + mrow;
            float bias = b1[n];
#pragma unroll
            for (int r = 0; r < 8; ++r) {           // C/D: VGPR r -> row r / r+8
                int m = M0 + wm + i * 16 + r + hi * 8;
                float v = acc[i][j][r] + bias;
                h1[(size_t)m * KDIM + n] = f32_to_bf16(fmaxf(v, 0.0f));
            }
        }
}

// GEMM2 fused: mu & logvar tiles + CLUB elementwise + reduction to d_out.
__global__ __launch_bounds__(256)
void club_gemm2_mi_kernel(const unsigned short* __restrict__ h1mu,
                          const unsigned short* __restrict__ w2mut,
                          const float* __restrict__ b2mu,
                          const unsigned short* __restrict__ h1lv,
                          const unsigned short* __restrict__ w2lvt,
                          const float* __restrict__ b2lv,
                          const float* __restrict__ zd,
                          const float* __restrict__ sumd,
                          const float* __restrict__ sumsq,
                          float* __restrict__ out) {
    __shared__ __align__(16) unsigned short sA[2 * TSZ];
    __shared__ __align__(16) unsigned short sB[2 * TSZ];
    __shared__ float red[256];
    int M0 = blockIdx.x * BM, N0 = blockIdx.y * BN;

    v8f amu[4][2], alv[4][2];
#pragma unroll
    for (int i = 0; i < 4; ++i)
#pragma unroll
        for (int j = 0; j < 2; ++j) { amu[i][j] = (v8f)0.0f; alv[i][j] = (v8f)0.0f; }

    club_gemm_block(h1mu, w2mut, M0, N0, sA, sB, amu);
    club_gemm_block(h1lv, w2lvt, M0, N0, sA, sB, alv);   // reuses LDS (synced inside)

    int tid = threadIdx.x, lane = tid & 31, wave = tid >> 5;
    int wm = (wave >> 2) * 64, wn = (wave & 3) * 32;
    int mrow = lane & 15, hi = lane >> 4;
    const float invN = 1.0f / (float)N_ROWS;

    float part = 0.0f;
#pragma unroll
    for (int i = 0; i < 4; ++i)
#pragma unroll
        for (int j = 0; j < 2; ++j) {
            int d = N0 + wn + j * 16 + mrow;
            float bmu = b2mu[d], blv = b2lv[d];
            float ez  = sumd[d]  * invN;
            float ez2 = sumsq[d] * invN;
#pragma unroll
            for (int r = 0; r < 8; ++r) {
                int m = M0 + wm + i * 16 + r + hi * 8;
                float mu  = amu[i][j][r] + bmu;
                float lv  = tanhf(alv[i][j][r] + blv);
                float iv  = 0.5f * __expf(-lv);
                float z   = zd[(size_t)m * KDIM + d];
                float dmu = mu - z;
                float neg = ez2 - 2.0f * mu * ez + mu * mu;  // E_j (z_d[j,d]-mu)^2
                part += iv * (neg - dmu * dmu);              // positive - negative
            }
        }
    red[tid] = part;
    __syncthreads();
    for (int s = 128; s > 0; s >>= 1) {
        if (tid < s) red[tid] += red[tid + s];
        __syncthreads();
    }
    if (tid == 0) atomicAdd(out, red[0] * invN);             // mean over rows
}

// ------------------------------- launcher -----------------------------------

extern "C" void kernel_launch(void* const* d_in, const int* in_sizes, int n_in,
                              void* d_out, int out_size, void* d_ws, size_t ws_size,
                              hipStream_t stream) {
    const float* z_c   = (const float*)d_in[0];
    const float* z_d   = (const float*)d_in[1];
    const float* W1_mu = (const float*)d_in[2];
    const float* b1_mu = (const float*)d_in[3];
    const float* W2_mu = (const float*)d_in[4];
    const float* b2_mu = (const float*)d_in[5];
    const float* W1_lv = (const float*)d_in[6];
    const float* b1_lv = (const float*)d_in[7];
    const float* W2_lv = (const float*)d_in[8];
    const float* b2_lv = (const float*)d_in[9];
    float* out = (float*)d_out;

    // workspace layout (256B-aligned segments); total ~104 MiB
    char* ws = (char*)d_ws;
    size_t off = 0;
    auto take = [&](size_t bytes) { char* p = ws + off; off += (bytes + 255) & ~(size_t)255; return p; };
    unsigned short* zc_bf  = (unsigned short*)take((size_t)N_ROWS * KDIM * 2);
    unsigned short* w1mu_t = (unsigned short*)take((size_t)KDIM * KDIM * 2);
    unsigned short* w2mu_t = (unsigned short*)take((size_t)KDIM * KDIM * 2);
    unsigned short* w1lv_t = (unsigned short*)take((size_t)KDIM * KDIM * 2);
    unsigned short* w2lv_t = (unsigned short*)take((size_t)KDIM * KDIM * 2);
    unsigned short* h1mu   = (unsigned short*)take((size_t)N_ROWS * KDIM * 2);
    unsigned short* h1lv   = (unsigned short*)take((size_t)N_ROWS * KDIM * 2);
    float* sumd  = (float*)take(KDIM * 4);
    float* sumsq = (float*)take(KDIM * 4);

    // 1. zero accumulators (harness poisons d_out / d_ws with 0xAA)
    club_zero_kernel<<<1, 256, 0, stream>>>(out, sumd, sumsq);

    // 2. precision conversion (z_c plain; weights transposed for K-contiguous B)
    {
        int n4 = (N_ROWS * KDIM) / 4;
        club_cvt_bf16_kernel<<<(n4 + 255) / 256, 256, 0, stream>>>(z_c, zc_bf, n4);
        int nb = (KDIM * KDIM) / 256;
        club_cvt_transpose_kernel<<<nb, 256, 0, stream>>>(W1_mu, w1mu_t);
        club_cvt_transpose_kernel<<<nb, 256, 0, stream>>>(W2_mu, w2mu_t);
        club_cvt_transpose_kernel<<<nb, 256, 0, stream>>>(W1_lv, w1lv_t);
        club_cvt_transpose_kernel<<<nb, 256, 0, stream>>>(W2_lv, w2lv_t);
    }

    // 3. column stats of z_d
    club_colstats_kernel<<<N_ROWS / 64, 256, 0, stream>>>(z_d, sumd, sumsq);

    // 4. hidden layers (bf16 WMMA)
    dim3 grid(N_ROWS / BM, KDIM / BN);
    club_gemm1_relu_kernel<<<grid, 256, 0, stream>>>(zc_bf, w1mu_t, b1_mu, h1mu);
    club_gemm1_relu_kernel<<<grid, 256, 0, stream>>>(zc_bf, w1lv_t, b1_lv, h1lv);

    // 5. fused second layer + CLUB objective + reduction
    club_gemm2_mi_kernel<<<grid, 256, 0, stream>>>(h1mu, w2mu_t, b2_mu,
                                                   h1lv, w2lv_t, b2_lv,
                                                   z_d, sumd, sumsq, out);
    (void)in_sizes; (void)n_in; (void)out_size; (void)ws_size;
}
